// GATC_67001489818013
// MI455X (gfx1250) — matrix operand order repaired and checked by reading, not compiled, not run on hardware
//
#include <hip/hip_runtime.h>

#define D_MODEL 128
#define EDIM    64
#define NHEAD   4
#define HC      32

typedef __attribute__((ext_vector_type(16))) __bf16       v16bf;
typedef __attribute__((ext_vector_type(8)))  float        v8f;
typedef __attribute__((ext_vector_type(8)))  unsigned int v8u;

static __device__ __forceinline__ unsigned short f2bf(float f) {
  unsigned u = __float_as_uint(f);
  u += 0x7FFFu + ((u >> 16) & 1u);          // round-to-nearest-even
  return (unsigned short)(u >> 16);
}
// order-preserving float -> uint mapping (for atomicMax-based segment max)
static __device__ __forceinline__ unsigned encf(float f) {
  unsigned u = __float_as_uint(f);
  return (u & 0x80000000u) ? ~u : (u | 0x80000000u);
}
static __device__ __forceinline__ float decf(unsigned u) {
  unsigned v = (u & 0x80000000u) ? (u & 0x7FFFFFFFu) : ~u;
  return __uint_as_float(v);
}

// ---------------------------------------------------------------- utilities
__global__ void fill_kernel(float* p, float v, long long n) {
  long long i = (long long)blockIdx.x * blockDim.x + threadIdx.x;
  if (i < n) p[i] = v;
}

// count in-degree and sum incoming edge_attr (for self-loop attr = mean)
__global__ void deg_loop_kernel(const long long* __restrict__ ei,
                                const float* __restrict__ ea,
                                float* __restrict__ deg,
                                float* __restrict__ loop_sum, int E) {
  long long tid = (long long)blockIdx.x * blockDim.x + threadIdx.x;
  if (tid >= (long long)E * EDIM) return;
  int e = (int)(tid >> 6);
  int c = (int)(tid & 63);
  long long d = ei[(long long)E + e];
  atomicAdd(&loop_sum[d * EDIM + c], ea[(long long)e * EDIM + c]);
  if (c == 0) atomicAdd(&deg[d], 1.0f);
}

__global__ void loop_div_kernel(float* loopat, const float* __restrict__ deg, int N) {
  long long tid = (long long)blockIdx.x * blockDim.x + threadIdx.x;
  if (tid >= (long long)N * EDIM) return;
  int n = (int)(tid >> 6);
  loopat[tid] /= fmaxf(deg[n], 1.0f);
}

// ---------------------------------------------------------------- WMMA GEMM
// Out[M x 128] = A[M x 128] @ W[128 x 128] (+ bias), fp32 in/out, bf16 WMMA.
// 128 threads = 4 waves; block computes a 64-row x 128-col tile.
// Wave w computes rows [w*16, w*16+16) x all 8 column tiles.
__global__ __launch_bounds__(128)
void gemm128_wmma(const float* __restrict__ A, const float* __restrict__ W,
                  const float* __restrict__ bias, float* __restrict__ Out, int M) {
  __shared__ unsigned short Ash[64 * 128];        // row-major bf16 bits
  __shared__ unsigned short Bsh[128 * 128];       // TRANSPOSED: Bsh[c*128 + k]
  const int t = threadIdx.x;
  const int rowBlk = blockIdx.x * 64;

  for (int i = t; i < 64 * 128; i += 128) {
    int r = i >> 7, c = i & 127;
    int gr = rowBlk + r;
    float v = (gr < M) ? A[(long long)gr * D_MODEL + c] : 0.0f;
    Ash[i] = f2bf(v);
  }
  for (int i = t; i < 128 * 128; i += 128) {
    int k = i >> 7, c = i & 127;
    Bsh[c * 128 + k] = f2bf(W[i]);
  }
  __syncthreads();

  const int lane = t & 31;
  const int half = lane >> 4;                     // 0: lanes 0-15, 1: lanes 16-31
  const int l16  = lane & 15;
  const int waveRow = (t >> 5) * 16;              // row offset within block
  const unsigned int* Au = (const unsigned int*)Ash;
  const unsigned int* Bu = (const unsigned int*)Bsh;

  v8f acc[8] = {};
  for (int kk = 0; kk < 4; ++kk) {                // K = 128 = 4 x 32
    const int kb = kk * 32;
    // A fragment (16x32 bf16): lane l16 = row; VGPR0-7 -> K = 16*half + 0..15
    // (same K order as B)
    const int arow = waveRow + l16;
    const int abase = (arow * 128 + kb + 16 * half) >> 1;   // uint index
    v8u av;
    #pragma unroll
    for (int i = 0; i < 4; ++i) av[i] = Au[abase + i];
    #pragma unroll
    for (int i = 0; i < 4; ++i) av[4 + i] = Au[abase + 4 + i];
    v16bf af = __builtin_bit_cast(v16bf, av);

    #pragma unroll
    for (int tile = 0; tile < 8; ++tile) {
      // B fragment (32x16): lane l16 = col; K = 16*half + 0..15
      const int col = tile * 16 + l16;
      const int bbase = (col * 128 + kb + 16 * half) >> 1;
      v8u bv;
      #pragma unroll
      for (int i = 0; i < 8; ++i) bv[i] = Bu[bbase + i];
      v16bf bf = __builtin_bit_cast(v16bf, bv);
      acc[tile] = __builtin_amdgcn_wmma_f32_16x16x32_bf16(
          false, af, false, bf, (short)0, acc[tile], false, false);
    }
  }
  // C/D layout: VGPR r -> M = r + 8*half, N = l16
  #pragma unroll
  for (int tile = 0; tile < 8; ++tile) {
    const int col = tile * 16 + l16;
    const float bv = bias ? bias[col] : 0.0f;
    #pragma unroll
    for (int r = 0; r < 8; ++r) {
      const int grow = rowBlk + waveRow + r + 8 * half;
      if (grow < M) Out[(long long)grow * D_MODEL + col] = acc[tile][r] + bv;
    }
  }
}

// ---------------------------------------------------------------- attention
// v_e[k][h] = sum_c We[k][h*32+c] * a_e[h][c]   (folds the Etot x 64 x 128 GEMM)
__global__ void ve_kernel(const float* __restrict__ We, const float* __restrict__ aeh,
                          float* __restrict__ ve) {
  int t = threadIdx.x;                            // 256 = 64*4
  int k = t >> 2, h = t & 3;
  float s = 0.0f;
  for (int c = 0; c < HC; ++c) s += We[k * D_MODEL + h * HC + c] * aeh[h * HC + c];
  ve[k * NHEAD + h] = s;
}

__global__ void alpha_nodes_kernel(const float* __restrict__ xh,
                                   const float* __restrict__ as_,
                                   const float* __restrict__ ad_,
                                   float* __restrict__ als, float* __restrict__ ald, int N) {
  long long tid = (long long)blockIdx.x * blockDim.x + threadIdx.x;
  if (tid >= (long long)N * NHEAD) return;
  long long n = tid >> 2;
  int h = (int)(tid & 3);
  float ss = 0.0f, sd = 0.0f;
  for (int c = 0; c < HC; ++c) {
    float v = xh[n * D_MODEL + h * HC + c];
    ss += v * as_[h * HC + c];
    sd += v * ad_[h * HC + c];
  }
  als[tid] = ss; ald[tid] = sd;
}

__global__ void alpha_e_kernel(const float* __restrict__ ea, const float* __restrict__ loopat,
                               const float* __restrict__ ve, float* __restrict__ aeo,
                               int E, int N) {
  long long tid = (long long)blockIdx.x * blockDim.x + threadIdx.x;
  if (tid >= (long long)(E + N) * NHEAD) return;
  long long idx = tid >> 2;
  int h = (int)(tid & 3);
  const float* ev = (idx < E) ? (ea + idx * EDIM) : (loopat + (idx - E) * EDIM);
  float s = 0.0f;
  for (int k = 0; k < EDIM; ++k) s += ev[k] * ve[k * NHEAD + h];
  aeo[tid] = s;
}

__global__ void pass1_kernel(const long long* __restrict__ ei,
                             const float* __restrict__ als, const float* __restrict__ ald,
                             const float* __restrict__ aee, unsigned* __restrict__ menc,
                             int E, int N) {
  long long tid = (long long)blockIdx.x * blockDim.x + threadIdx.x;
  if (tid >= (long long)(E + N) * NHEAD) return;
  long long e = tid >> 2;
  int h = (int)(tid & 3);
  long long s, d;
  if (e < E) { s = ei[e]; d = ei[(long long)E + e]; } else { s = d = e - E; }
  float lg = als[s * 4 + h] + ald[d * 4 + h] + aee[tid];
  lg = lg > 0.0f ? lg : 0.2f * lg;                // leaky_relu(0.2)
  atomicMax(&menc[d * 4 + h], encf(lg));
}

// ex = exp(logit - max); accumulate sum(ex) and sum(ex * xh[src]) into agg[dst]
// one lane handles 4 channels; att normalization deferred to epilogue.
__global__ void pass2_kernel(const long long* __restrict__ ei,
                             const float* __restrict__ als, const float* __restrict__ ald,
                             const float* __restrict__ aee, const unsigned* __restrict__ menc,
                             float* __restrict__ sden, const float* __restrict__ xh,
                             float* __restrict__ agg, int E, int N) {
  long long tid = (long long)blockIdx.x * blockDim.x + threadIdx.x;
  if (tid >= (long long)(E + N) * 32) return;
  long long e = tid >> 5;
  int lane = (int)(tid & 31);
  int c0 = lane * 4;
  int h = lane >> 3;                              // 4 channels all inside one head
  long long s, d;
  if (e < E) { s = ei[e]; d = ei[(long long)E + e]; } else { s = d = e - E; }
  float lg = als[s * 4 + h] + ald[d * 4 + h] + aee[e * 4 + h];
  lg = lg > 0.0f ? lg : 0.2f * lg;
  float ex = __expf(lg - decf(menc[d * 4 + h]));
  if ((lane & 7) == 0) atomicAdd(&sden[d * 4 + h], ex);
  const float4 xv = *(const float4*)(xh + s * D_MODEL + c0);
  float* ap = agg + d * D_MODEL + c0;
  atomicAdd(ap + 0, ex * xv.x);
  atomicAdd(ap + 1, ex * xv.y);
  atomicAdd(ap + 2, ex * xv.z);
  atomicAdd(ap + 3, ex * xv.w);
}

// out = relu(h + agg/(s*cnt) + b);  cnt = deg+1 (self-loop included)
__global__ void pass3_kernel(const float* __restrict__ hin, const float* __restrict__ agg,
                             const float* __restrict__ sden, const float* __restrict__ deg,
                             const float* __restrict__ bias, float* __restrict__ hout, int N) {
  long long tid = (long long)blockIdx.x * blockDim.x + threadIdx.x;
  if (tid >= (long long)N * D_MODEL) return;
  long long n = tid >> 7;
  int c = (int)(tid & 127);
  int h = c >> 5;
  float cnt = deg[n] + 1.0f;
  float o = agg[tid] / (sden[n * 4 + h] * cnt) + bias[c];
  float v = hin[tid] + o;
  hout[tid] = v > 0.0f ? v : 0.0f;
}

// ---------------------------------------------------------------- launcher
static inline unsigned nblk(long long n, int b) { return (unsigned)((n + b - 1) / b); }

extern "C" void kernel_launch(void* const* d_in, const int* in_sizes, int n_in,
                              void* d_out, int out_size, void* d_ws, size_t ws_size,
                              hipStream_t stream) {
  const float*     x   = (const float*)d_in[0];
  const long long* ei  = (const long long*)d_in[1];  // jnp.int64 per reference
  const float*     ea  = (const float*)d_in[2];
  const float*     lpW = (const float*)d_in[3];
  const float*     lpb = (const float*)d_in[4];
  const int N = in_sizes[0] / D_MODEL;
  const int E = in_sizes[1] / 2;
  const long long Etot = (long long)E + N;

  float* ws = (float*)d_ws;
  size_t off = 0;
  float* hA     = ws + off; off += (size_t)N * D_MODEL;
  float* hB     = ws + off; off += (size_t)N * D_MODEL;
  float* xh     = ws + off; off += (size_t)N * D_MODEL;
  float* agg    = ws + off; off += (size_t)N * D_MODEL;
  float* loopat = ws + off; off += (size_t)N * EDIM;
  float* deg    = ws + off; off += (size_t)N;
  float* als    = ws + off; off += (size_t)N * NHEAD;
  float* ald    = ws + off; off += (size_t)N * NHEAD;
  float* sden   = ws + off; off += (size_t)N * NHEAD;
  unsigned* menc = (unsigned*)(ws + off); off += (size_t)N * NHEAD;
  float* aee    = ws + off; off += (size_t)Etot * NHEAD;
  float* ve     = ws + off; off += (size_t)EDIM * NHEAD;
  (void)ws_size; (void)n_in; (void)out_size;

  // self-loop attrs (mean of incoming) + degree
  fill_kernel<<<nblk(N, 256), 256, 0, stream>>>(deg, 0.0f, (long long)N);
  fill_kernel<<<nblk((long long)N * EDIM, 256), 256, 0, stream>>>(loopat, 0.0f, (long long)N * EDIM);
  deg_loop_kernel<<<nblk((long long)E * EDIM, 256), 256, 0, stream>>>(ei, ea, deg, loopat, E);
  loop_div_kernel<<<nblk((long long)N * EDIM, 256), 256, 0, stream>>>(loopat, deg, N);

  // input projection: h = x @ lp_W + lp_b
  gemm128_wmma<<<dim3((N + 63) / 64), 128, 0, stream>>>(x, lpW, lpb, hA, N);

  const float* hcur = hA;
  for (int l = 0; l < 3; ++l) {
    const float* W   = (const float*)d_in[5 + l * 6 + 0];
    const float* as_ = (const float*)d_in[5 + l * 6 + 1];
    const float* ad_ = (const float*)d_in[5 + l * 6 + 2];
    const float* We  = (const float*)d_in[5 + l * 6 + 3];
    const float* aeh = (const float*)d_in[5 + l * 6 + 4];
    const float* bb  = (const float*)d_in[5 + l * 6 + 5];
    float* hout = (l == 0) ? hB : (l == 1) ? hA : (float*)d_out;

    ve_kernel<<<1, 256, 0, stream>>>(We, aeh, ve);
    gemm128_wmma<<<dim3((N + 63) / 64), 128, 0, stream>>>(hcur, W, nullptr, xh, N);
    alpha_nodes_kernel<<<nblk((long long)N * NHEAD, 256), 256, 0, stream>>>(xh, as_, ad_, als, ald, N);
    alpha_e_kernel<<<nblk(Etot * NHEAD, 256), 256, 0, stream>>>(ea, loopat, ve, aee, E, N);

    fill_kernel<<<nblk((long long)N * NHEAD, 256), 256, 0, stream>>>((float*)menc, 0.0f, (long long)N * NHEAD);
    fill_kernel<<<nblk((long long)N * NHEAD, 256), 256, 0, stream>>>(sden, 0.0f, (long long)N * NHEAD);
    fill_kernel<<<nblk((long long)N * D_MODEL, 256), 256, 0, stream>>>(agg, 0.0f, (long long)N * D_MODEL);

    pass1_kernel<<<nblk(Etot * NHEAD, 256), 256, 0, stream>>>(ei, als, ald, aee, menc, E, N);
    pass2_kernel<<<nblk(Etot * 32, 256), 256, 0, stream>>>(ei, als, ald, aee, menc, sden, xh, agg, E, N);
    pass3_kernel<<<nblk((long long)N * D_MODEL, 256), 256, 0, stream>>>(hcur, agg, sden, deg, bb, hout, N);
    hcur = hout;
  }
}
